// QRNN_44633300140403
// MI455X (gfx1250) — 8-bit matrix operand order repaired and checked by reading, compiled, not run on hardware
//
#include <hip/hip_runtime.h>
#include <hip/hip_bf16.h>

typedef __attribute__((ext_vector_type(8))) int v8i;

#define DEV __device__ __forceinline__

constexpr int BB = 64, SS = 256, EE = 512, HH = 512, OO = 2, VV = 32000;
constexpr int MM = SS * BB;

constexpr size_t OFF_SLOTS = 0;
constexpr size_t OFF_HS    = 1024;
constexpr size_t OFF_WIH0  = 8192;
constexpr size_t OFF_WHH0  = OFF_WIH0 + (size_t)2 * HH * EE;
constexpr size_t OFF_WIH1  = OFF_WHH0 + (size_t)2 * HH * HH;
constexpr size_t OFF_WHH1  = OFF_WIH1 + (size_t)2 * HH * 2 * HH;
constexpr size_t OFF_FIN   = OFF_WHH1 + (size_t)2 * HH * HH;
constexpr size_t OFF_X     = (size_t)4 << 20;
constexpr size_t OFF_YS0   = OFF_X + (size_t)MM * EE;
constexpr size_t OFF_XW0   = OFF_YS0 + (size_t)2 * MM * HH;
constexpr size_t OFF_XW1   = OFF_XW0;


DEV float scale_from(unsigned bits) {
  float m = __uint_as_float(bits) + 1e-12f;
  return exp2f(ceilf(log2f(m * (1.0f / 127.0f))));
}
DEV float fxp_q(float x, float s) {
  float r = rintf(x / s);
  r = fminf(fmaxf(r, -128.0f), 127.0f);
  return r * s;
}
DEV int fxp_i8(float x, float s) {
  float r = rintf(x / s);
  r = fminf(fmaxf(r, -128.0f), 127.0f);
  return (int)r;
}

DEV v8i load_a_i8(const signed char* A, int lda, int lane) {
  const int m = lane & 15, g = (lane >> 4) & 1;
  const signed char* row = A + (size_t)m * lda + g * 8;
  v8i a;
#pragma unroll
  for (int p = 0; p < 4; ++p) {
    const int2 d = *(const int2*)(row + p * 16);
    a[2 * p] = d.x; a[2 * p + 1] = d.y;
  }
  return a;
}
DEV v8i load_b_i8(const signed char* W, int ldb, int lane) {
  const int n = lane & 15, g = (lane >> 4) & 1;
  const signed char* col = W + (size_t)n * ldb + g * 8;
  v8i b;
#pragma unroll
  for (int p = 0; p < 4; ++p) {
    const int2 d = *(const int2*)(col + p * 16);
    b[2 * p] = d.x; b[2 * p + 1] = d.y;
  }
  return b;
}

__global__ void init_slots_kernel(unsigned* slots) { slots[threadIdx.x] = 0u; }

__global__ void absmax_kernel(const float* __restrict__ x, int n, unsigned* __restrict__ slot) {
  __shared__ unsigned smax;
  if (threadIdx.x == 0) smax = 0u;
  __syncthreads();
  float m = 0.0f;
  for (int i = blockIdx.x * blockDim.x + threadIdx.x; i < n; i += gridDim.x * blockDim.x)
    m = fmaxf(m, fabsf(x[i]));
  atomicMax(&smax, __float_as_uint(m));
  __syncthreads();
  if (threadIdx.x == 0) atomicMax(slot, smax);
}

__global__ void quant_i8_kernel(const float* __restrict__ src, signed char* __restrict__ dst,
                                int n, const unsigned* __restrict__ slot) {
  const float s = scale_from(*slot);
  for (int i = blockIdx.x * blockDim.x + threadIdx.x; i < n; i += gridDim.x * blockDim.x)
    dst[i] = (signed char)fxp_i8(src[i], s);
}

__global__ void gather_q_kernel(const int* __restrict__ text, const float* __restrict__ emb,
                                signed char* __restrict__ x8,
                                const unsigned* __restrict__ emb_slot,
                                unsigned* __restrict__ x_slot) {
  __shared__ unsigned smax;
  if (threadIdx.x == 0) smax = 0u;
  __syncthreads();
  const float se = scale_from(*emb_slot);
  const int n = MM * EE;
  float m = 0.0f;
  for (int i = blockIdx.x * blockDim.x + threadIdx.x; i < n; i += gridDim.x * blockDim.x) {
    const int e = i & (EE - 1);
    const int sb = i >> 9;
    const int s = sb >> 6, b = sb & 63;
    const int tok = text[b * SS + s];
    const int q = fxp_i8(emb[(size_t)tok * EE + e], se);
    x8[i] = (signed char)q;
    m = fmaxf(m, fabsf((float)q * se));
  }
  atomicMax(&smax, __float_as_uint(m));
  __syncthreads();
  if (threadIdx.x == 0) atomicMax(x_slot, smax);
}

__global__ void requant_kernel(signed char* __restrict__ x8,
                               const unsigned* __restrict__ emb_slot,
                               const unsigned* __restrict__ x_slot, int n) {
  const float se = scale_from(*emb_slot);
  const float sx = scale_from(*x_slot);
  for (int i = blockIdx.x * blockDim.x + threadIdx.x; i < n; i += gridDim.x * blockDim.x)
    x8[i] = (signed char)fxp_i8((float)x8[i] * se, sx);
}

__global__ __launch_bounds__(128)
void proj0_kernel(const signed char* __restrict__ x, const signed char* __restrict__ wih,
                  const float* __restrict__ bih, const float* __restrict__ bhh,
                  const unsigned* __restrict__ slots, float* __restrict__ xw) {
  const int lane = threadIdx.x & 31, wave = threadIdx.x >> 5;
  const int d = blockIdx.z, mt = blockIdx.x, nt = blockIdx.y * 4 + wave;
  const signed char* A = x + (size_t)mt * 16 * EE;
  const signed char* W = wih + (size_t)d * HH * EE + (size_t)nt * 16 * EE;
  v8i acc = {};
#pragma unroll
  for (int kc = 0; kc < 8; ++kc) {
    v8i a = load_a_i8(A + kc * 64, EE, lane);
    v8i b = load_b_i8(W + kc * 64, EE, lane);
    acc = __builtin_amdgcn_wmma_i32_16x16x64_iu8(true, a, true, b, acc, false, false);
  }
  const float sc = scale_from(slots[1]) * scale_from(slots[2 + d]);
  const int n = nt * 16 + (lane & 15);
  const int g8 = (lane >> 4) * 8;
  const float bias = bih[d * HH + n] + bhh[d * HH + n];
  float* outp = xw + (size_t)d * MM * HH;
#pragma unroll
  for (int r = 0; r < 8; ++r)
    outp[(size_t)(mt * 16 + g8 + r) * HH + n] = bias + sc * (float)acc[r];
}

__global__ __launch_bounds__(128)
void proj1_kernel(const signed char* __restrict__ ys, const signed char* __restrict__ wih,
                  const float* __restrict__ bih, const float* __restrict__ bhh,
                  const unsigned* __restrict__ slots, const float* __restrict__ hs,
                  float* __restrict__ xw) {
  const int lane = threadIdx.x & 31, wave = threadIdx.x >> 5;
  const int d = blockIdx.z, mt = blockIdx.x, nt = blockIdx.y * 4 + wave;
  const int t = mt >> 2;
  const signed char* A0 = ys + (size_t)mt * 16 * HH;
  const signed char* A1 = ys + (size_t)MM * HH + (size_t)mt * 16 * HH;
  const signed char* W = wih + (size_t)d * HH * 2 * HH + (size_t)nt * 16 * 2 * HH;
  v8i acc0 = {}, acc1 = {};
#pragma unroll
  for (int kc = 0; kc < 8; ++kc) {
    v8i a0 = load_a_i8(A0 + kc * 64, HH, lane);
    v8i b0 = load_b_i8(W + kc * 64, 2 * HH, lane);
    acc0 = __builtin_amdgcn_wmma_i32_16x16x64_iu8(true, a0, true, b0, acc0, false, false);
    v8i a1 = load_a_i8(A1 + kc * 64, HH, lane);
    v8i b1 = load_b_i8(W + HH + kc * 64, 2 * HH, lane);
    acc1 = __builtin_amdgcn_wmma_i32_16x16x64_iu8(true, a1, true, b1, acc1, false, false);
  }
  const float sw = scale_from(slots[6 + d]);
  const float s0 = hs[t] * sw, s1 = hs[SS + t] * sw;
  const int n = nt * 16 + (lane & 15);
  const int g8 = (lane >> 4) * 8;
  const float bias = bih[d * HH + n] + bhh[d * HH + n];
  float* outp = xw + (size_t)d * MM * HH;
#pragma unroll
  for (int r = 0; r < 8; ++r)
    outp[(size_t)(mt * 16 + g8 + r) * HH + n] = bias + s0 * (float)acc0[r] + s1 * (float)acc1[r];
}

__global__ __launch_bounds__(512)
void rec_kernel(const float* __restrict__ xw, const signed char* __restrict__ whh,
                const unsigned* __restrict__ whh_slots, float* __restrict__ hs,
                signed char* __restrict__ ys, float* __restrict__ finals, int write_ys) {
  const int d = blockIdx.x;
  const int lane = threadIdx.x & 31, wave = threadIdx.x >> 5;
  const int g8 = (lane >> 4) * 8;
  const int nl = lane & 15;
  const float* xwd = xw + (size_t)d * MM * HH;
  const signed char* W = whh + (size_t)d * HH * HH;
  signed char* ysd = write_ys ? ys + (size_t)d * MM * HH : nullptr;
  float* find = finals + (size_t)d * BB * HH;
  float* hsrow = hs + d * SS;
  const float swhh = scale_from(whh_slots[d]);

  const int nt0 = 2 * wave, nt1 = 2 * wave + 1;
  const int n0 = nt0 * 16 + nl, n1 = nt1 * 16 + nl;

  v8i Bf0[8], Bf1[8];
#pragma unroll
  for (int kc = 0; kc < 8; ++kc) {
    Bf0[kc] = load_b_i8(W + (size_t)nt0 * 16 * HH + kc * 64, HH, lane);
    Bf1[kc] = load_b_i8(W + (size_t)nt1 * 16 * HH + kc * 64, HH, lane);
  }

  __shared__ signed char h8[BB * HH];
  __shared__ unsigned sred[2];

  for (int i = threadIdx.x; i < BB * HH / 16; i += 512)
    ((int4*)h8)[i] = make_int4(0, 0, 0, 0);
  float hsc = 1.0f;
  __syncthreads();

  for (int ts = 0; ts < SS; ++ts) {
    const int t = d ? (SS - 1 - ts) : ts;
    if (threadIdx.x == 0) { sred[0] = 0u; sred[1] = 0u; }
    __syncthreads();
    const float sc = hsc * swhh;

    float pre0[4][8], pre1[4][8];
    float lm = 0.0f;
#pragma unroll
    for (int mt = 0; mt < 4; ++mt) {
      v8i acc0 = {}, acc1 = {};
      const signed char* Ab = h8 + (size_t)mt * 16 * HH;
#pragma unroll
      for (int kc = 0; kc < 8; ++kc) {
        v8i a = load_a_i8(Ab + kc * 64, HH, lane);
        acc0 = __builtin_amdgcn_wmma_i32_16x16x64_iu8(true, a, true, Bf0[kc], acc0, false, false);
        acc1 = __builtin_amdgcn_wmma_i32_16x16x64_iu8(true, a, true, Bf1[kc], acc1, false, false);
      }
      const float* xrow = xwd + ((size_t)t * BB + mt * 16 + g8) * HH;
#pragma unroll
      for (int r = 0; r < 8; ++r) {
        const float p0 = xrow[(size_t)r * HH + n0] + sc * (float)acc0[r];
        const float p1 = xrow[(size_t)r * HH + n1] + sc * (float)acc1[r];
        pre0[mt][r] = p0;
        pre1[mt][r] = p1;
        lm = fmaxf(lm, fmaxf(fabsf(p0), fabsf(p1)));
      }
    }
    atomicMax(&sred[0], __float_as_uint(lm));
    __syncthreads();
    const float s1 = scale_from(sred[0]);

    float lm2 = 0.0f;
#pragma unroll
    for (int mt = 0; mt < 4; ++mt)
#pragma unroll
      for (int r = 0; r < 8; ++r) {
        const float h0 = tanhf(fxp_q(pre0[mt][r], s1));
        const float h1 = tanhf(fxp_q(pre1[mt][r], s1));
        pre0[mt][r] = h0;
        pre1[mt][r] = h1;
        lm2 = fmaxf(lm2, fmaxf(fabsf(h0), fabsf(h1)));
      }
    atomicMax(&sred[1], __float_as_uint(lm2));
    __syncthreads();
    const float s2 = scale_from(sred[1]);

#pragma unroll
    for (int mt = 0; mt < 4; ++mt)
#pragma unroll
      for (int r = 0; r < 8; ++r) {
        const int m = mt * 16 + g8 + r;
        const int q0 = fxp_i8(pre0[mt][r], s2);
        const int q1 = fxp_i8(pre1[mt][r], s2);
        h8[(size_t)m * HH + n0] = (signed char)q0;
        h8[(size_t)m * HH + n1] = (signed char)q1;
        if (write_ys) {
          ysd[((size_t)t * BB + m) * HH + n0] = (signed char)q0;
          ysd[((size_t)t * BB + m) * HH + n1] = (signed char)q1;
        }
        if (ts == SS - 1) {
          find[(size_t)m * HH + n0] = (float)q0 * s2;
          find[(size_t)m * HH + n1] = (float)q1 * s2;
        }
      }
    if (threadIdx.x == 0) hsrow[t] = s2;
    hsc = s2;
    __syncthreads();
  }
}

__global__ __launch_bounds__(512)
void fc_kernel(const float* __restrict__ finals, const float* __restrict__ fcw,
               const unsigned* __restrict__ slots, float* __restrict__ out) {
  const int tid = threadIdx.x;
  const int o = tid & 1, b = (tid >> 1) & 63, j = tid >> 7;
  const float s = scale_from(slots[10]);
  float acc = 0.0f;
  for (int k = 0; k < HH; ++k)
    acc += finals[((size_t)j * BB + b) * HH + k] * fxp_q(fcw[o * HH + k], s);
  out[((size_t)j * BB + b) * OO + o] = acc;
}

extern "C" void kernel_launch(void* const* d_in, const int* in_sizes, int n_in,
                              void* d_out, int out_size, void* d_ws, size_t ws_size,
                              hipStream_t stream) {
  (void)in_sizes; (void)n_in; (void)out_size; (void)ws_size;
  const int*   text = (const int*)d_in[0];
  const float* emb  = (const float*)d_in[2];
  const float* wih0 = (const float*)d_in[3];
  const float* whh0 = (const float*)d_in[4];
  const float* bih0 = (const float*)d_in[5];
  const float* bhh0 = (const float*)d_in[6];
  const float* wih1 = (const float*)d_in[7];
  const float* whh1 = (const float*)d_in[8];
  const float* bih1 = (const float*)d_in[9];
  const float* bhh1 = (const float*)d_in[10];
  const float* fcw  = (const float*)d_in[11];

  char* ws = (char*)d_ws;
  unsigned*    slots  = (unsigned*)(ws + OFF_SLOTS);
  float*       hs     = (float*)(ws + OFF_HS);
  signed char* q_wih0 = (signed char*)(ws + OFF_WIH0);
  signed char* q_whh0 = (signed char*)(ws + OFF_WHH0);
  signed char* q_wih1 = (signed char*)(ws + OFF_WIH1);
  signed char* q_whh1 = (signed char*)(ws + OFF_WHH1);
  float*       fin    = (float*)(ws + OFF_FIN);
  signed char* x8     = (signed char*)(ws + OFF_X);
  signed char* ys0    = (signed char*)(ws + OFF_YS0);
  float*       xw0    = (float*)(ws + OFF_XW0);
  float*       xw1    = (float*)(ws + OFF_XW1);

  init_slots_kernel<<<1, 64, 0, stream>>>(slots);

  absmax_kernel<<<512, 256, 0, stream>>>(emb, VV * EE, slots + 0);
  for (int d = 0; d < 2; ++d) {
    absmax_kernel<<<128, 256, 0, stream>>>(wih0 + (size_t)d * HH * EE, HH * EE, slots + 2 + d);
    absmax_kernel<<<128, 256, 0, stream>>>(whh0 + (size_t)d * HH * HH, HH * HH, slots + 4 + d);
    absmax_kernel<<<256, 256, 0, stream>>>(wih1 + (size_t)d * HH * 2 * HH, HH * 2 * HH, slots + 6 + d);
    absmax_kernel<<<128, 256, 0, stream>>>(whh1 + (size_t)d * HH * HH, HH * HH, slots + 8 + d);
  }
  absmax_kernel<<<1, 256, 0, stream>>>(fcw, OO * HH, slots + 10);

  for (int d = 0; d < 2; ++d) {
    quant_i8_kernel<<<1024, 256, 0, stream>>>(wih0 + (size_t)d * HH * EE,
                                              q_wih0 + (size_t)d * HH * EE, HH * EE, slots + 2 + d);
    quant_i8_kernel<<<1024, 256, 0, stream>>>(whh0 + (size_t)d * HH * HH,
                                              q_whh0 + (size_t)d * HH * HH, HH * HH, slots + 4 + d);
    quant_i8_kernel<<<2048, 256, 0, stream>>>(wih1 + (size_t)d * HH * 2 * HH,
                                              q_wih1 + (size_t)d * HH * 2 * HH, HH * 2 * HH, slots + 6 + d);
    quant_i8_kernel<<<1024, 256, 0, stream>>>(whh1 + (size_t)d * HH * HH,
                                              q_whh1 + (size_t)d * HH * HH, HH * HH, slots + 8 + d);
  }

  gather_q_kernel<<<8192, 256, 0, stream>>>(text, emb, x8, slots + 0, slots + 1);
  requant_kernel<<<8192, 256, 0, stream>>>(x8, slots + 0, slots + 1, MM * EE);

  proj0_kernel<<<dim3(MM / 16, HH / 64, 2), 128, 0, stream>>>(x8, q_wih0, bih0, bhh0, slots, xw0);
  rec_kernel<<<2, 512, 0, stream>>>(xw0, q_whh0, slots + 4, hs, ys0, fin, 1);
  proj1_kernel<<<dim3(MM / 16, HH / 64, 2), 128, 0, stream>>>(ys0, q_wih1, bih1, bhh1, slots, hs, xw1);
  rec_kernel<<<2, 512, 0, stream>>>(xw1, q_whh1, slots + 8, hs + 2 * SS, nullptr,
                                    fin + (size_t)2 * BB * HH, 0);
  fc_kernel<<<1, 512, 0, stream>>>(fin, fcw, slots, (float*)d_out);
}
